// OmniGenTransformer_29515015258322
// MI455X (gfx1250) — matrix operand order repaired and checked by reading, not compiled, not run on hardware
//
#include <hip/hip_runtime.h>
#include <hip/hip_bf16.h>

#define Bsz  2
#define Ssz  1024
#define Dsz  2048
#define Hsz  32
#define HDsz 64
#define Isz  8192
#define Msz  (Bsz * Ssz)   // 2048 rows

typedef __bf16 bf16_t;
typedef __attribute__((ext_vector_type(16))) __bf16 v16bf;
typedef __attribute__((ext_vector_type(8)))  __bf16 v8bf;
typedef __attribute__((ext_vector_type(4)))  __bf16 v4bf;
typedef __attribute__((ext_vector_type(8)))  float  v8f;
typedef __attribute__((ext_vector_type(4)))  float  v4f;

// ---------------------------------------------------------------------------
// CDNA5 async global->LDS copy (GLOBAL_LOAD_ASYNC_TO_LDS_B128, ASYNCcnt).
// VDST = per-lane LDS byte address, VADDR = 64-bit global address (GV mode).
// LDS byte offset is the low 32 bits of the generic pointer (LDS aperture).
// ---------------------------------------------------------------------------
__device__ __forceinline__ void async_copy_b128(void* lds_ptr, const void* gptr) {
    unsigned lds_off = (unsigned)(unsigned long long)lds_ptr;
    asm volatile("global_load_async_to_lds_b128 %0, %1, off"
                 :: "v"(lds_off), "v"(gptr)
                 : "memory");
}
__device__ __forceinline__ void wait_async0() {
    asm volatile("s_wait_asynccnt 0" ::: "memory");
}

// ---------------------------------------------------------------------------
// WMMA fragment loaders (wave32, 16-bit operands, per CDNA5 ISA 7.12.2)
// A: 16(m) x 32(k). Lane l: m = l&15, kh = l>>4.
//    VGPR 0..3 pack k = kh*8 + {0..7}; VGPR 4..7 pack k = 16 + kh*8 + {0..7}.
// B is consumed from an N-major tile [n][k]; lane l: n = l&15, holds
//    k = kh*8 + {0..7} and 16 + kh*8 + {0..7}  (two 16B chunks each).
// ---------------------------------------------------------------------------
__device__ __forceinline__ v16bf load_frag_a(const bf16_t* base, int ld) {
    const int lane = threadIdx.x & 31;
    const int m  = lane & 15;
    const int kh = lane >> 4;
    const bf16_t* p = base + (size_t)m * ld;
    v8bf lo = *(const v8bf*)(p + kh * 8);
    v8bf hi = *(const v8bf*)(p + 16 + kh * 8);
    v16bf r;
#pragma unroll
    for (int i = 0; i < 8; ++i) { r[i] = lo[i]; r[8 + i] = hi[i]; }
    return r;
}

__device__ __forceinline__ v16bf load_frag_b(const bf16_t* base, int ld) {
    const int lane = threadIdx.x & 31;
    const int n  = lane & 15;
    const int kh = lane >> 4;
    const bf16_t* p = base + (size_t)n * ld;
    v8bf lo = *(const v8bf*)(p + kh * 8);
    v8bf hi = *(const v8bf*)(p + 16 + kh * 8);
    v16bf r;
#pragma unroll
    for (int i = 0; i < 8; ++i) { r[i] = lo[i]; r[8 + i] = hi[i]; }
    return r;
}

__device__ __forceinline__ v8f wmma_bf16(v16bf a, v16bf b, v8f c) {
    return __builtin_amdgcn_wmma_f32_16x16x32_bf16(false, a, false, b,
                                                   (short)0, c, false, false);
}

// ---------------------------------------------------------------------------
// LayerNorm (f32 in) -> bf16 out. One 256-thread block per row of D=2048.
// ---------------------------------------------------------------------------
__global__ __launch_bounds__(256) void layernorm_bf16_kernel(
    const float* __restrict__ X, const float* __restrict__ g,
    const float* __restrict__ b, bf16_t* __restrict__ Y, int D)
{
    const int row  = blockIdx.x;
    const int tid  = threadIdx.x;
    const int lane = tid & 31, wave = tid >> 5;
    const float* x = X + (size_t)row * D;

    float s = 0.f, ss = 0.f;
    for (int i = tid * 4; i < D; i += 256 * 4) {
        v4f v = *(const v4f*)(x + i);
#pragma unroll
        for (int j = 0; j < 4; ++j) { s += v[j]; ss += v[j] * v[j]; }
    }
#pragma unroll
    for (int o = 16; o > 0; o >>= 1) { s += __shfl_xor(s, o); ss += __shfl_xor(ss, o); }

    __shared__ float red[16];
    if (lane == 0) { red[wave] = s; red[8 + wave] = ss; }
    __syncthreads();
    if (wave == 0) {
        float a = (lane < 8) ? red[lane] : 0.f;
        float c = (lane < 8) ? red[8 + lane] : 0.f;
#pragma unroll
        for (int o = 4; o > 0; o >>= 1) { a += __shfl_xor(a, o); c += __shfl_xor(c, o); }
        if (lane == 0) { red[0] = a; red[8] = c; }
    }
    __syncthreads();
    const float mu   = red[0] / (float)D;
    const float var  = red[8] / (float)D - mu * mu;
    const float rsig = rsqrtf(var + 1e-5f);

    bf16_t* y = Y + (size_t)row * D;
    for (int i = tid * 4; i < D; i += 256 * 4) {
        v4f v  = *(const v4f*)(x + i);
        v4f gg = *(const v4f*)(g + i);
        v4f bb = *(const v4f*)(b + i);
        v4bf o;
#pragma unroll
        for (int j = 0; j < 4; ++j)
            o[j] = (bf16_t)((v[j] - mu) * rsig * gg[j] + bb[j]);
        *(v4bf*)(y + i) = o;
    }
}

// ---------------------------------------------------------------------------
// Weight transpose + f32->bf16 convert: W[K][N] f32  ->  Wt[N][K] bf16
// ---------------------------------------------------------------------------
__global__ __launch_bounds__(256) void transpose_bf16_kernel(
    const float* __restrict__ W, bf16_t* __restrict__ Wt, int K, int N)
{
    __shared__ float t[32][33];
    const int bx = blockIdx.x * 32;   // N
    const int by = blockIdx.y * 32;   // K
    const int tx = threadIdx.x & 31, ty = threadIdx.x >> 5;
#pragma unroll
    for (int i = 0; i < 32; i += 8)
        t[ty + i][tx] = W[(size_t)(by + ty + i) * N + bx + tx];
    __syncthreads();
#pragma unroll
    for (int i = 0; i < 32; i += 8)
        Wt[(size_t)(bx + ty + i) * K + by + tx] = (bf16_t)t[tx][ty + i];
}

// ---------------------------------------------------------------------------
// BF16 WMMA GEMM:  C[M][N] (f32) = A[M][K](bf16) * Bt[N][K](bf16)^T
// Block tile 128x128x64, 8 waves (2x4), each wave 64x32 via 4x2 WMMA tiles.
// Tiles staged with double-buffered GLOBAL_LOAD_ASYNC_TO_LDS_B128 copies:
// issue tile k+1, run 16 WMMAs on tile k, s_wait_asynccnt 0, barrier, swap.
// BIAS: add bias[N].  RES: C += result (residual accumulate).
// ---------------------------------------------------------------------------
template <bool BIAS, bool RES>
__global__ __launch_bounds__(256) void gemm_bf16_kernel(
    const bf16_t* __restrict__ A, const bf16_t* __restrict__ Bt,
    const float* __restrict__ bias, float* __restrict__ C,
    int M, int N, int K)
{
    constexpr int BK = 64;
    __shared__ bf16_t sA[2][128][BK + 8];
    __shared__ bf16_t sB[2][128][BK + 8];

    const int tid  = threadIdx.x;
    const int lane = tid & 31;
    const int wave = tid >> 5;
    const int wm   = wave >> 2;          // 0..1
    const int wn   = wave & 3;           // 0..3
    const int bm   = blockIdx.y * 128;
    const int bn   = blockIdx.x * 128;

    v8f acc[4][2] = {};

    // stage one 128x64 A tile + 128x64 B tile into LDS buffer `buf`
    auto stage = [&](int buf, int k0) {
#pragma unroll
        for (int i = 0; i < 4; ++i) {
            int id = tid + i * 256;
            int r = id >> 3, c = (id & 7) * 8;
            async_copy_b128(&sA[buf][r][c], A + (size_t)(bm + r) * K + k0 + c);
            async_copy_b128(&sB[buf][r][c], Bt + (size_t)(bn + r) * K + k0 + c);
        }
    };

    stage(0, 0);
    wait_async0();
    __syncthreads();

    int cur = 0;
    for (int k0 = 0; k0 < K; k0 += BK) {
        if (k0 + BK < K) stage(cur ^ 1, k0 + BK);   // overlap with compute

#pragma unroll
        for (int kk = 0; kk < BK; kk += 32) {
            v16bf bf[2];
#pragma unroll
            for (int j = 0; j < 2; ++j)
                bf[j] = load_frag_b(&sB[cur][wn * 32 + j * 16][kk], BK + 8);
#pragma unroll
            for (int i = 0; i < 4; ++i) {
                v16bf af = load_frag_a(&sA[cur][wm * 64 + i * 16][kk], BK + 8);
#pragma unroll
                for (int j = 0; j < 2; ++j)
                    acc[i][j] = wmma_bf16(af, bf[j], acc[i][j]);
            }
        }
        wait_async0();      // next tile's async copies landed in LDS
        __syncthreads();    // everyone done reading `cur`, next buffer ready
        cur ^= 1;
    }

    // epilogue: C-tile layout -> row m = r + 8*(lane>>4), col n = lane&15
    const int ln = lane & 15, lh = lane >> 4;
#pragma unroll
    for (int i = 0; i < 4; ++i) {
#pragma unroll
        for (int j = 0; j < 2; ++j) {
            const int col = bn + wn * 32 + j * 16 + ln;
            const float bv = BIAS ? bias[col] : 0.0f;
#pragma unroll
            for (int r = 0; r < 8; ++r) {
                const int row = bm + wm * 64 + i * 16 + lh * 8 + r;
                float v = acc[i][j][r] + bv;
                float* cp = C + (size_t)row * N + col;
                if (RES) *cp = *cp + v; else *cp = v;
            }
        }
    }
}

// ---------------------------------------------------------------------------
// Flash-style attention with WMMA for QK^T and P*V.
// Grid: B * H * (S/128). Block: 256 threads (8 waves); wave owns 16 q-rows.
// qkv layout: [B*S][3*D], q at col h*64, k at D + h*64, v at 2D + h*64.
// Writes ctx as bf16 [B*S][D].
// ---------------------------------------------------------------------------
__global__ __launch_bounds__(256) void attention_kernel(
    const float* __restrict__ qkv, const int* __restrict__ mask,
    bf16_t* __restrict__ ctx)
{
    constexpr int AQ = 128, AJ = 32, LD3 = 3 * Dsz;
    int t = blockIdx.x;
    const int qb = t % (Ssz / AQ); t /= (Ssz / AQ);
    const int h  = t % Hsz;
    const int b  = t / Hsz;

    __shared__ bf16_t sQ[AQ][HDsz + 8];
    __shared__ bf16_t sK[AJ][HDsz + 8];
    __shared__ bf16_t sV[HDsz][AJ + 16];
    __shared__ bf16_t sP[8][16][32];
    __shared__ float  sMB[AJ];

    const int tid = threadIdx.x, lane = tid & 31, wave = tid >> 5;
    const int ln = lane & 15, lh = lane >> 4;
    const size_t row0 = (size_t)b * Ssz + qb * AQ;
    const int qcol = h * HDsz, kcol = Dsz + h * HDsz, vcol = 2 * Dsz + h * HDsz;

    // stage Q (128 x 64) as bf16
    for (int i = tid; i < AQ * (HDsz / 4); i += 256) {
        const int r = i / 16, c = (i % 16) * 4;
        v4f v = *(const v4f*)(qkv + (row0 + r) * LD3 + qcol + c);
        v4bf o;
#pragma unroll
        for (int j = 0; j < 4; ++j) o[j] = (bf16_t)v[j];
        *(v4bf*)&sQ[r][c] = o;
    }

    float m_i[8], l_i[8];
#pragma unroll
    for (int r = 0; r < 8; ++r) { m_i[r] = -1e30f; l_i[r] = 0.f; }
    v8f oacc[4] = {};
    const float scale = 0.125f;   // 1/sqrt(64)

    for (int j0 = 0; j0 < Ssz; j0 += AJ) {
        __syncthreads();
        // K tile (32 x 64) row-major bf16
        for (int i = tid; i < AJ * 16; i += 256) {
            const int r = i / 16, c = (i % 16) * 4;
            v4f v = *(const v4f*)(qkv + ((size_t)b * Ssz + j0 + r) * LD3 + kcol + c);
            v4bf o;
#pragma unroll
            for (int j = 0; j < 4; ++j) o[j] = (bf16_t)v[j];
            *(v4bf*)&sK[r][c] = o;
        }
        // V tile transposed: sV[hd][key]
        for (int i = tid; i < AJ * 16; i += 256) {
            const int r = i / 16, c = (i % 16) * 4;
            v4f v = *(const v4f*)(qkv + ((size_t)b * Ssz + j0 + r) * LD3 + vcol + c);
#pragma unroll
            for (int j = 0; j < 4; ++j) sV[c + j][r] = (bf16_t)v[j];
        }
        if (tid < AJ)
            sMB[tid] = (1.0f - (float)mask[(size_t)b * Ssz + j0 + tid]) * -10000.0f;
        __syncthreads();

        // scores: 16 q-rows x 32 keys  (2 k-steps over HD, 2 n-tiles)
        v8f sacc[2] = {};
#pragma unroll
        for (int ks = 0; ks < 2; ++ks) {
            v16bf aq = load_frag_a(&sQ[wave * 16][ks * 32], HDsz + 8);
#pragma unroll
            for (int j = 0; j < 2; ++j) {
                v16bf bk = load_frag_b(&sK[j * 16][ks * 32], HDsz + 8);
                sacc[j] = wmma_bf16(aq, bk, sacc[j]);
            }
        }

        // online softmax per row (row m = lh*8 + r, 16 lanes hold its n-vals)
        const float mb0 = sMB[ln], mb1 = sMB[16 + ln];
        float alpha[8];
#pragma unroll
        for (int r = 0; r < 8; ++r) {
            float s0 = sacc[0][r] * scale + mb0;
            float s1 = sacc[1][r] * scale + mb1;
            float mx = fmaxf(s0, s1);
#pragma unroll
            for (int o = 8; o > 0; o >>= 1) mx = fmaxf(mx, __shfl_xor(mx, o, 16));
            const float mnew = fmaxf(m_i[r], mx);
            const float a  = __expf(m_i[r] - mnew);
            const float p0 = __expf(s0 - mnew);
            const float p1 = __expf(s1 - mnew);
            float rs = p0 + p1;
#pragma unroll
            for (int o = 8; o > 0; o >>= 1) rs += __shfl_xor(rs, o, 16);
            l_i[r] = l_i[r] * a + rs;
            m_i[r] = mnew;
            alpha[r] = a;
            sP[wave][lh * 8 + r][ln]      = (bf16_t)p0;
            sP[wave][lh * 8 + r][16 + ln] = (bf16_t)p1;
        }
#pragma unroll
        for (int tt = 0; tt < 4; ++tt)
#pragma unroll
            for (int r = 0; r < 8; ++r) oacc[tt][r] *= alpha[r];
        __syncthreads();

        // ctx += P(16x32) * V(32x64)
        v16bf ap = load_frag_a(&sP[wave][0][0], 32);
#pragma unroll
        for (int tt = 0; tt < 4; ++tt) {
            v16bf bv = load_frag_b(&sV[tt * 16][0], AJ + 16);
            oacc[tt] = wmma_bf16(ap, bv, oacc[tt]);
        }
    }

    // normalize and write ctx (bf16)
#pragma unroll
    for (int tt = 0; tt < 4; ++tt)
#pragma unroll
        for (int r = 0; r < 8; ++r) {
            const int qrow = qb * AQ + wave * 16 + lh * 8 + r;
            const int col  = h * HDsz + tt * 16 + ln;
            ctx[((size_t)b * Ssz + qrow) * Dsz + col] = (bf16_t)(oacc[tt][r] / l_i[r]);
        }
}

// ---------------------------------------------------------------------------
// SwiGLU: act[r][i] = silu(gu[r][i]) * gu[r][I+i], output bf16
// ---------------------------------------------------------------------------
__global__ __launch_bounds__(256) void swiglu_kernel(
    const float* __restrict__ gu, bf16_t* __restrict__ out, int rows, int icols)
{
    const size_t chunks = (size_t)rows * icols / 4;
    const int ic4 = icols / 4;
    for (size_t c = (size_t)blockIdx.x * blockDim.x + threadIdx.x; c < chunks;
         c += (size_t)gridDim.x * blockDim.x) {
        const size_t r = c / ic4;
        const int cc = (int)(c % ic4) * 4;
        const float* gp = gu + r * (size_t)(2 * icols) + cc;
        v4f g = *(const v4f*)gp;
        v4f u = *(const v4f*)(gp + icols);
        v4bf o;
#pragma unroll
        for (int j = 0; j < 4; ++j) {
            float sg = g[j] / (1.0f + __expf(-g[j]));
            o[j] = (bf16_t)(sg * u[j]);
        }
        *(v4bf*)(out + r * (size_t)icols + cc) = o;
    }
}

// ---------------------------------------------------------------------------
extern "C" void kernel_launch(void* const* d_in, const int* in_sizes, int n_in,
                              void* d_out, int out_size, void* d_ws, size_t ws_size,
                              hipStream_t stream) {
    const float* hs     = (const float*)d_in[0];
    const int*   amask  = (const int*)d_in[1];
    const float* ln1_g  = (const float*)d_in[2];
    const float* ln1_b  = (const float*)d_in[3];
    const float* qkv_w  = (const float*)d_in[4];
    const float* o_w    = (const float*)d_in[5];
    const float* ln2_g  = (const float*)d_in[6];
    const float* ln2_b  = (const float*)d_in[7];
    const float* gu_w   = (const float*)d_in[8];
    const float* gu_b   = (const float*)d_in[9];
    const float* down_w = (const float*)d_in[10];
    const float* down_b = (const float*)d_in[11];
    float* h = (float*)d_out;   // running hidden state [M][D]

    // h = hidden_states
    hipMemcpyAsync(h, hs, sizeof(float) * (size_t)Msz * Dsz,
                   hipMemcpyDeviceToDevice, stream);

    // workspace carve
    size_t off = 0;
    auto carve = [&](size_t bytes) {
        void* p = (char*)d_ws + off;
        off += (bytes + 255) & ~(size_t)255;
        return p;
    };
    bf16_t* wT   = (bf16_t*)carve((size_t)Dsz * 2 * Isz * sizeof(bf16_t)); // 67 MB max
    bf16_t* xb   = (bf16_t*)carve((size_t)Msz * Dsz * sizeof(bf16_t));     // LN out / ctx
    float*  qkvb = (float*) carve((size_t)Msz * 3 * Dsz * sizeof(float));
    float*  gub  = (float*) carve((size_t)Msz * 2 * Isz * sizeof(float));
    bf16_t* actb = (bf16_t*)carve((size_t)Msz * Isz * sizeof(bf16_t));
    (void)ws_size; (void)n_in; (void)in_sizes; (void)out_size;

    const dim3 blk(256);
    for (int l = 0; l < 2; ++l) {
        const float* qw = qkv_w  + (size_t)l * Dsz * 3 * Dsz;
        const float* ow = o_w    + (size_t)l * Dsz * Dsz;
        const float* gw = gu_w   + (size_t)l * Dsz * 2 * Isz;
        const float* gb = gu_b   + (size_t)l * 2 * Isz;
        const float* dw = down_w + (size_t)l * Isz * Dsz;
        const float* db = down_b + (size_t)l * Dsz;

        // x = LN1(h)  -> bf16
        layernorm_bf16_kernel<<<Msz, blk, 0, stream>>>(h, ln1_g + l * Dsz, ln1_b + l * Dsz, xb, Dsz);
        // qkv = x @ qkv_w
        transpose_bf16_kernel<<<dim3(3 * Dsz / 32, Dsz / 32), blk, 0, stream>>>(qw, wT, Dsz, 3 * Dsz);
        gemm_bf16_kernel<false, false><<<dim3(3 * Dsz / 128, Msz / 128), blk, 0, stream>>>(
            xb, wT, nullptr, qkvb, Msz, 3 * Dsz, Dsz);
        // attention -> ctx (bf16, reuse xb)
        attention_kernel<<<Bsz * Hsz * (Ssz / 128), blk, 0, stream>>>(qkvb, amask, xb);
        // h += ctx @ o_w
        transpose_bf16_kernel<<<dim3(Dsz / 32, Dsz / 32), blk, 0, stream>>>(ow, wT, Dsz, Dsz);
        gemm_bf16_kernel<false, true><<<dim3(Dsz / 128, Msz / 128), blk, 0, stream>>>(
            xb, wT, nullptr, h, Msz, Dsz, Dsz);
        // y = LN2(h) -> bf16
        layernorm_bf16_kernel<<<Msz, blk, 0, stream>>>(h, ln2_g + l * Dsz, ln2_b + l * Dsz, xb, Dsz);
        // gu = y @ gu_w + gu_b
        transpose_bf16_kernel<<<dim3(2 * Isz / 32, Dsz / 32), blk, 0, stream>>>(gw, wT, Dsz, 2 * Isz);
        gemm_bf16_kernel<true, false><<<dim3(2 * Isz / 128, Msz / 128), blk, 0, stream>>>(
            xb, wT, gb, gub, Msz, 2 * Isz, Dsz);
        // act = silu(gate) * up -> bf16
        swiglu_kernel<<<4096, blk, 0, stream>>>(gub, actb, Msz, Isz);
        // h += act @ down_w + down_b
        transpose_bf16_kernel<<<dim3(Dsz / 32, Isz / 32), blk, 0, stream>>>(dw, wT, Isz, Dsz);
        gemm_bf16_kernel<true, true><<<dim3(Dsz / 128, Msz / 128), blk, 0, stream>>>(
            actb, wT, db, h, Msz, Dsz, Isz);
    }
}
